// VIT_23021024707221
// MI455X (gfx1250) — matrix operand order repaired and checked by reading, not compiled, not run on hardware
//
#include <hip/hip_runtime.h>
#include <hip/hip_bf16.h>

// ---------------------------------------------------------------------------
// Types for CDNA5 WMMA (wave32, 16x16x32 bf16 -> f32 accumulate)
// ---------------------------------------------------------------------------
typedef __attribute__((ext_vector_type(16))) __bf16          v16bf;
typedef __attribute__((ext_vector_type(16))) unsigned short  v16u;
typedef __attribute__((ext_vector_type(8)))  unsigned short  v8u;
typedef __attribute__((ext_vector_type(8)))  float           v8f;
typedef __attribute__((__vector_size__(4 * sizeof(int)))) int vit_i4;  // matches builtin param

// CDNA5 async global->LDS copy (ASYNCcnt) if the toolchain exposes it.
#if defined(__AMDGCN__) && __has_builtin(__builtin_amdgcn_global_load_async_to_lds_b128) && __has_builtin(__builtin_amdgcn_s_wait_asynccnt)
#define VIT_HAVE_ASYNC 1
#else
#define VIT_HAVE_ASYNC 0
#endif

__device__ __forceinline__ unsigned short f2bf(float f) {
    union { float f; unsigned u; } c; c.f = f;
    unsigned r = 0x7FFFu + ((c.u >> 16) & 1u);      // round-to-nearest-even
    return (unsigned short)((c.u + r) >> 16);
}

__device__ __forceinline__ v8f vzero() {
    v8f z;
#pragma unroll
    for (int i = 0; i < 8; ++i) z[i] = 0.0f;
    return z;
}

__device__ __forceinline__ v8u uzero() {
    v8u z;
#pragma unroll
    for (int i = 0; i < 8; ++i) z[i] = 0;
    return z;
}

// 16-byte global->LDS copy: async DMA when available, else vector ld/st.
__device__ __forceinline__ void copy16_g2l(const unsigned short* __restrict__ g,
                                           unsigned short* __restrict__ l) {
#if VIT_HAVE_ASYNC
    __builtin_amdgcn_global_load_async_to_lds_b128((vit_i4*)g, (vit_i4*)l, 0, 0);
#else
    *(v8u*)l = *(const v8u*)g;
#endif
}

__device__ __forceinline__ void async_fence() {
#if VIT_HAVE_ASYNC
    __builtin_amdgcn_s_wait_asynccnt(0);
#endif
}

// Assemble a 16-element bf16 fragment from two contiguous 8-element LDS loads.
__device__ __forceinline__ v16bf mkfrag(v8u lo, v8u hi) {
    v16u t;
#pragma unroll
    for (int i = 0; i < 8; ++i) { t[i] = lo[i]; t[i + 8] = hi[i]; }
    return __builtin_bit_cast(v16bf, t);
}

// ---------------------------------------------------------------------------
// Generic bf16 WMMA GEMM:  out[M,N] = A[M,K] * B[K,N] (+bias)(+relu)(+residual)
// Block: 256 threads (8 waves). Tile: BM=128, BN=64, BK=32, double-buffered.
// Wave w computes rows [w*16, w*16+16) x 4 column tiles of 16.
// LDS A: [128][32] row-major (async-copied);  LDS Bt: [64][32] (N-major,
// transposed on store) so both A- and B-frags are contiguous b128 LDS loads.
// Requires M%128==0, N%64==0, K%32==0 (true for all calls here).
// ---------------------------------------------------------------------------
__global__ __launch_bounds__(256)
void vit_gemm_bf16_wmma(const unsigned short* __restrict__ A,
                        const unsigned short* __restrict__ B,
                        const float* __restrict__ bias,      // nullable
                        const float* __restrict__ residual,  // nullable
                        float* __restrict__ outF,            // nullable
                        unsigned short* __restrict__ outB,   // nullable
                        int M, int N, int K, int relu)
{
    __shared__ __align__(16) unsigned short ldsA[2][128 * 32];
    __shared__ __align__(16) unsigned short ldsBt[2][64 * 32];

    const int tid  = threadIdx.x;
    const int lane = tid & 31;
    const int w    = tid >> 5;
    const int hf   = lane >> 4;     // which 16-lane half of the wave
    const int r16  = lane & 15;
    const int m0   = blockIdx.x * 128;
    const int n0   = blockIdx.y * 64;

    // Staging geometry (all 16B chunks):
    //  A tile 128x32 = 512 chunks -> 2 per thread;  B tile 32x64 = 256 -> 1.
    const int am    = tid >> 2;            // A chunk row (t=0); t=1 adds 64
    const int akoff = (tid & 3) * 8;       // A chunk K offset
    const int bk    = tid >> 3;            // B chunk K row
    const int bnoff = (tid & 7) * 8;       // B chunk N offset

    v8f acc[4];
#pragma unroll
    for (int i = 0; i < 4; ++i) acc[i] = vzero();

    const int ksteps = K >> 5;

    auto stage = [&](int buf, int kk) {
        // A: straight contiguous copy -> async DMA to LDS
#pragma unroll
        for (int t = 0; t < 2; ++t) {
            int m = t * 64 + am;
            copy16_g2l(A + (size_t)(m0 + m) * K + kk + akoff,
                       &ldsA[buf][m * 32 + akoff]);
        }
        // B: 16B vector load + 8-way transpose scatter into [n][k]
        v8u bv = *(const v8u*)(B + (size_t)(kk + bk) * N + n0 + bnoff);
#pragma unroll
        for (int j = 0; j < 8; ++j)
            ldsBt[buf][(bnoff + j) * 32 + bk] = bv[j];
    };

    stage(0, 0);
    async_fence();
    __syncthreads();

    for (int s = 0; s < ksteps; ++s) {
        const int cur = s & 1;
        if (s + 1 < ksteps) stage(cur ^ 1, (s + 1) << 5);   // prefetch next tile

        // A fragment for this wave (16 rows x 32 K)
        const unsigned short* pa = &ldsA[cur][(w * 16 + r16) * 32];
        v16bf af = mkfrag(*(const v8u*)(pa + 16 * hf),
                          *(const v8u*)(pa + 16 * hf + 8));
#pragma unroll
        for (int nt = 0; nt < 4; ++nt) {
            const unsigned short* pb = &ldsBt[cur][(nt * 16 + r16) * 32 + 16 * hf];
            v16bf bf = mkfrag(*(const v8u*)pb, *(const v8u*)(pb + 8));
            acc[nt] = __builtin_amdgcn_wmma_f32_16x16x32_bf16(
                false, af, false, bf, (short)0, acc[nt], false, false);
        }
        async_fence();
        __syncthreads();
    }

    // Epilogue: C/D layout -> lane r16 = col, VGPR rr -> row rr + 8*hf
#pragma unroll
    for (int nt = 0; nt < 4; ++nt) {
        int col = n0 + nt * 16 + r16;
        float bv = bias ? bias[col] : 0.0f;
#pragma unroll
        for (int rr = 0; rr < 8; ++rr) {
            int row = m0 + w * 16 + rr + 8 * hf;
            float v = acc[nt][rr] + bv;
            if (relu) v = v > 0.0f ? v : 0.0f;
            size_t o = (size_t)row * N + col;
            if (residual) v += residual[o];
            if (outF) outF[o] = v;
            if (outB) outB[o] = f2bf(v);
        }
    }
}

// ---------------------------------------------------------------------------
// Fused attention: one block per (batch, head).  T=65 padded to 80 rows,
// K padded to 96 for the P@V contraction.  All-WMMA for QK^T and P@V.
// 128 threads = 4 waves.  LDS: Q/P overlay 80x96, K 80x64, V^T 64x96, S 80x84.
// ---------------------------------------------------------------------------
__global__ __launch_bounds__(128)
void vit_attention_wmma(const unsigned short* __restrict__ Q,
                        const unsigned short* __restrict__ Km,
                        const unsigned short* __restrict__ V,
                        unsigned short* __restrict__ O)
{
    __shared__ __align__(16) unsigned short ldsQP[80 * 96]; // Q, then P (bf16)
    __shared__ __align__(16) unsigned short ldsK[80 * 64];
    __shared__ __align__(16) unsigned short ldsVt[64 * 96]; // [d][s]
    __shared__ __align__(16) float          sS[80 * 84];

    const int bh = blockIdx.x;            // b*12 + h
    const int b = bh / 12, h = bh % 12;
    const int tid = threadIdx.x, lane = tid & 31, w = tid >> 5;
    const int hf = lane >> 4, r16 = lane & 15;
    const size_t base = ((size_t)b * 65) * 768 + (size_t)h * 64;

    // Q (stride 96) and K (stride 64): 16B chunks, async for valid rows
    for (int c = tid; c < 80 * 8; c += 128) {
        int t = c >> 3, doff = (c & 7) * 8;
        if (t < 65) {
            copy16_g2l(Q  + base + (size_t)t * 768 + doff, &ldsQP[t * 96 + doff]);
            copy16_g2l(Km + base + (size_t)t * 768 + doff, &ldsK[t * 64 + doff]);
        } else {
            *(v8u*)&ldsQP[t * 96 + doff] = uzero();
            *(v8u*)&ldsK[t * 64 + doff]  = uzero();
        }
    }
    // V transposed: vector load along d, scatter into ldsVt[d][s]
    for (int c = tid; c < 80 * 8; c += 128) {
        int s = c >> 3, doff = (c & 7) * 8;
        v8u vv = (s < 65) ? *(const v8u*)(V + base + (size_t)s * 768 + doff)
                          : uzero();
#pragma unroll
        for (int j = 0; j < 8; ++j) ldsVt[(doff + j) * 96 + s] = vv[j];
    }
    for (int i = tid; i < 64 * 16; i += 128) {          // zero s = 80..95
        int d = i >> 4, s = 80 + (i & 15);
        ldsVt[d * 96 + s] = 0;
    }
    async_fence();
    __syncthreads();

    const float scale = 0.03608439182435161f;   // 768^-0.5 (reference uses E^-0.5)

    // S = Q @ K^T  (80x80, 25 tiles round-robined over 4 waves; wave-uniform)
    for (int tile = w; tile < 25; tile += 4) {
        int mt = tile / 5, nt = tile % 5;
        v8f acc = vzero();
#pragma unroll
        for (int kk = 0; kk < 64; kk += 32) {
            const unsigned short* pa = &ldsQP[(mt * 16 + r16) * 96 + kk];
            v16bf af = mkfrag(*(const v8u*)(pa + 16 * hf),
                              *(const v8u*)(pa + 16 * hf + 8));
            const unsigned short* pb = &ldsK[(nt * 16 + r16) * 64 + kk + 16 * hf];
            v16bf bf = mkfrag(*(const v8u*)pb, *(const v8u*)(pb + 8));
            acc = __builtin_amdgcn_wmma_f32_16x16x32_bf16(
                false, af, false, bf, (short)0, acc, false, false);
        }
#pragma unroll
        for (int rr = 0; rr < 8; ++rr)
            sS[(mt * 16 + rr + 8 * hf) * 84 + nt * 16 + r16] = acc[rr] * scale;
    }
    __syncthreads();

    // Row softmax over valid 65 columns; write bf16 P into the (dead) Q buffer
    if (tid < 80) {
        if (tid < 65) {
            float* rowp = &sS[tid * 84];
            float m = -1e30f;
            for (int s = 0; s < 65; ++s) m = fmaxf(m, rowp[s]);
            float sum = 0.0f;
            for (int s = 0; s < 65; ++s) { float e = __expf(rowp[s] - m); rowp[s] = e; sum += e; }
            float inv = 1.0f / sum;
            for (int s = 0; s < 65; ++s) ldsQP[tid * 96 + s] = f2bf(rowp[s] * inv);
            for (int s = 65; s < 96; ++s) ldsQP[tid * 96 + s] = 0;
        } else {
            for (int s = 0; s < 96; ++s) ldsQP[tid * 96 + s] = 0;
        }
    }
    __syncthreads();

    // O = P @ V  (80x64, K padded to 96; 20 tiles over 4 waves)
    for (int tile = w; tile < 20; tile += 4) {
        int mt = tile / 4, nt = tile % 4;
        v8f acc = vzero();
#pragma unroll
        for (int kk = 0; kk < 96; kk += 32) {
            const unsigned short* pa = &ldsQP[(mt * 16 + r16) * 96 + kk];
            v16bf af = mkfrag(*(const v8u*)(pa + 16 * hf),
                              *(const v8u*)(pa + 16 * hf + 8));
            const unsigned short* pb = &ldsVt[(nt * 16 + r16) * 96 + kk + 16 * hf];
            v16bf bf = mkfrag(*(const v8u*)pb, *(const v8u*)(pb + 8));
            acc = __builtin_amdgcn_wmma_f32_16x16x32_bf16(
                false, af, false, bf, (short)0, acc, false, false);
        }
#pragma unroll
        for (int rr = 0; rr < 8; ++rr) {
            int t = mt * 16 + rr + 8 * hf;
            if (t < 65)
                O[base + (size_t)t * 768 + nt * 16 + r16] = f2bf(acc[rr]);
        }
    }
}

// ---------------------------------------------------------------------------
// Elementwise / reduction helper kernels
// ---------------------------------------------------------------------------
__global__ void vit_f32_to_bf16(const float* __restrict__ in,
                                unsigned short* __restrict__ out, long n)
{
    long i = (long)blockIdx.x * 256 + threadIdx.x;
    if (i < n) out[i] = f2bf(in[i]);
}

__global__ void vit_patchify(const float* __restrict__ img,
                             unsigned short* __restrict__ Pm)
{
    int idx = blockIdx.x * 256 + threadIdx.x;     // 8192*768 elements
    if (idx >= 8192 * 768) return;
    int row = idx / 768, col = idx % 768;
    int b = row >> 6, pidx = row & 63, py = pidx >> 3, px = pidx & 7;
    int c = col % 3, t = col / 3, iy = t >> 4, ix = t & 15;
    size_t src = (((size_t)(b * 128 + py * 16 + iy)) * 128 + (px * 16 + ix)) * 3 + c;
    Pm[idx] = f2bf(img[src]);
}

__global__ void vit_assemble_x(const float* __restrict__ emb,
                               const float* __restrict__ cls,
                               const float* __restrict__ pos,
                               float* __restrict__ x)
{
    int idx = blockIdx.x * 256 + threadIdx.x;     // 128*65*768 elements
    if (idx >= 128 * 65 * 768) return;
    int e = idx % 768; int bt = idx / 768; int t = bt % 65; int b = bt / 65;
    float v = (t == 0) ? cls[e] : emb[((size_t)b * 64 + (t - 1)) * 768 + e];
    x[idx] = v + pos[t * 768 + e];
}

__global__ __launch_bounds__(256)
void vit_layernorm_bf16(const float* __restrict__ x, const float* __restrict__ g,
                        const float* __restrict__ bb, unsigned short* __restrict__ out)
{
    const int E = 768;
    int row = blockIdx.x;
    const float* xr = x + (size_t)row * E;
    __shared__ float red[256];
    float s = 0.0f;
    for (int i = threadIdx.x; i < E; i += 256) s += xr[i];
    red[threadIdx.x] = s; __syncthreads();
    for (int st = 128; st > 0; st >>= 1) {
        if (threadIdx.x < st) red[threadIdx.x] += red[threadIdx.x + st];
        __syncthreads();
    }
    float mu = red[0] / E; __syncthreads();
    float vs = 0.0f;
    for (int i = threadIdx.x; i < E; i += 256) { float d = xr[i] - mu; vs += d * d; }
    red[threadIdx.x] = vs; __syncthreads();
    for (int st = 128; st > 0; st >>= 1) {
        if (threadIdx.x < st) red[threadIdx.x] += red[threadIdx.x + st];
        __syncthreads();
    }
    float rstd = rsqrtf(red[0] / E + 1e-5f);
    for (int i = threadIdx.x; i < E; i += 256)
        out[(size_t)row * E + i] = f2bf((xr[i] - mu) * rstd * g[i] + bb[i]);
}

__global__ void vit_gather_cls(const float* __restrict__ x,
                               unsigned short* __restrict__ out)
{
    int i = blockIdx.x * 256 + threadIdx.x;       // 128*768
    if (i >= 128 * 768) return;
    int b = i / 768, e = i % 768;
    out[i] = f2bf(x[((size_t)b * 65) * 768 + e]);
}

__global__ __launch_bounds__(256)
void vit_softmax256(const float* __restrict__ in, float* __restrict__ out)
{
    int b = blockIdx.x;
    __shared__ float red[256];
    float v = in[b * 256 + threadIdx.x];
    red[threadIdx.x] = v; __syncthreads();
    for (int st = 128; st > 0; st >>= 1) {
        if (threadIdx.x < st) red[threadIdx.x] = fmaxf(red[threadIdx.x], red[threadIdx.x + st]);
        __syncthreads();
    }
    float m = red[0]; __syncthreads();
    float e = __expf(v - m);
    red[threadIdx.x] = e; __syncthreads();
    for (int st = 128; st > 0; st >>= 1) {
        if (threadIdx.x < st) red[threadIdx.x] += red[threadIdx.x + st];
        __syncthreads();
    }
    out[b * 256 + threadIdx.x] = e / red[0];
}

// ---------------------------------------------------------------------------
// Host orchestration
// ---------------------------------------------------------------------------
static inline size_t align256(size_t x) { return (x + 255) & ~(size_t)255; }

extern "C" void kernel_launch(void* const* d_in, const int* in_sizes, int n_in,
                              void* d_out, int out_size, void* d_ws, size_t ws_size,
                              hipStream_t stream)
{
    const float* images = (const float*)d_in[0];
    const float* lin_w  = (const float*)d_in[1];
    const float* cls    = (const float*)d_in[2];
    const float* pos    = (const float*)d_in[3];
    const float* Wq     = (const float*)d_in[4];
    const float* Wk     = (const float*)d_in[5];
    const float* Wv     = (const float*)d_in[6];
    const float* Wo     = (const float*)d_in[7];
    const float* bo     = (const float*)d_in[8];
    const float* ln1g   = (const float*)d_in[9];
    const float* ln1b   = (const float*)d_in[10];
    const float* ln2g   = (const float*)d_in[11];
    const float* ln2b   = (const float*)d_in[12];
    const float* W1     = (const float*)d_in[13];
    const float* b1     = (const float*)d_in[14];
    const float* W2     = (const float*)d_in[15];
    const float* b2     = (const float*)d_in[16];
    const float* head_w = (const float*)d_in[17];
    const float* head_b = (const float*)d_in[18];

    const int L = 12, E = 768, T = 65, Bn = 128, NT = Bn * T; // 8320 rows

    // Carve workspace
    char* ws = (char*)d_ws;
    size_t off = 0;
    auto carve = [&](size_t bytes) -> char* {
        char* p = ws + off; off = align256(off + bytes); return p;
    };
    unsigned short* linB = (unsigned short*)carve((size_t)E * E * 2);
    unsigned short* WqB  = (unsigned short*)carve((size_t)L * E * E * 2);
    unsigned short* WkB  = (unsigned short*)carve((size_t)L * E * E * 2);
    unsigned short* WvB  = (unsigned short*)carve((size_t)L * E * E * 2);
    unsigned short* WoB  = (unsigned short*)carve((size_t)L * E * E * 2);
    unsigned short* W1B  = (unsigned short*)carve((size_t)L * E * 4 * E * 2);
    unsigned short* W2B  = (unsigned short*)carve((size_t)L * 4 * E * E * 2);
    unsigned short* WhB  = (unsigned short*)carve((size_t)E * 256 * 2);
    unsigned short* Pm   = (unsigned short*)carve((size_t)8192 * E * 2);
    float*          emb  = (float*)        carve((size_t)8192 * E * 4);
    float*          x    = (float*)        carve((size_t)NT * E * 4);
    unsigned short* hB   = (unsigned short*)carve((size_t)NT * E * 2);
    unsigned short* QB   = (unsigned short*)carve((size_t)NT * E * 2);
    unsigned short* KB   = (unsigned short*)carve((size_t)NT * E * 2);
    unsigned short* VB   = (unsigned short*)carve((size_t)NT * E * 2);
    unsigned short* OB   = (unsigned short*)carve((size_t)NT * E * 2);
    unsigned short* HidB = (unsigned short*)carve((size_t)NT * 4 * E * 2);
    unsigned short* clsB = (unsigned short*)carve((size_t)Bn * E * 2);
    float*          lg   = (float*)        carve((size_t)Bn * 256 * 4);
    (void)ws_size; (void)n_in; (void)in_sizes; (void)out_size;

    auto conv = [&](const float* in, unsigned short* out, long n) {
        vit_f32_to_bf16<<<dim3((unsigned)((n + 255) / 256)), 256, 0, stream>>>(in, out, n);
    };
    conv(lin_w,  linB, (long)E * E);
    conv(Wq,     WqB,  (long)L * E * E);
    conv(Wk,     WkB,  (long)L * E * E);
    conv(Wv,     WvB,  (long)L * E * E);
    conv(Wo,     WoB,  (long)L * E * E);
    conv(W1,     W1B,  (long)L * E * 4 * E);
    conv(W2,     W2B,  (long)L * 4 * E * E);
    conv(head_w, WhB,  (long)E * 256);

    auto gemm = [&](const unsigned short* A, const unsigned short* Bm,
                    const float* bias, const float* res,
                    float* oF, unsigned short* oB,
                    int M, int N, int K, int relu) {
        dim3 g(M / 128, N / 64);
        vit_gemm_bf16_wmma<<<g, 256, 0, stream>>>(A, Bm, bias, res, oF, oB, M, N, K, relu);
    };

    // Patch embedding: x = [cls; patches @ lin_w] + pos
    vit_patchify<<<(8192 * 768 + 255) / 256, 256, 0, stream>>>(images, Pm);
    gemm(Pm, linB, nullptr, nullptr, emb, nullptr, 8192, E, E, 0);
    vit_assemble_x<<<(128 * 65 * 768 + 255) / 256, 256, 0, stream>>>(emb, cls, pos, x);

    for (int l = 0; l < L; ++l) {
        size_t wo  = (size_t)l * E * E;
        size_t wo1 = (size_t)l * E * 4 * E;
        // h = LN1(x);  Q,K,V = h @ Wq/Wk/Wv
        vit_layernorm_bf16<<<NT, 256, 0, stream>>>(x, ln1g + l * E, ln1b + l * E, hB);
        gemm(hB, WqB + wo, nullptr, nullptr, nullptr, QB, NT, E, E, 0);
        gemm(hB, WkB + wo, nullptr, nullptr, nullptr, KB, NT, E, E, 0);
        gemm(hB, WvB + wo, nullptr, nullptr, nullptr, VB, NT, E, E, 0);
        // attn = softmax(QK^T * E^-0.5) V   (per batch*head)
        vit_attention_wmma<<<Bn * 12, 128, 0, stream>>>(QB, KB, VB, OB);
        // x = x + attn @ Wo + bo
        gemm(OB, WoB + wo, bo + l * E, x, x, nullptr, NT, E, E, 0);
        // x = x + relu(LN2(x) @ W1 + b1) @ W2 + b2
        vit_layernorm_bf16<<<NT, 256, 0, stream>>>(x, ln2g + l * E, ln2b + l * E, hB);
        gemm(hB, W1B + wo1, b1 + l * 4 * E, nullptr, nullptr, HidB, NT, 4 * E, E, 1);
        gemm(HidB, W2B + wo1, b2 + l * E, x, x, nullptr, NT, E, 4 * E, 0);
    }

    // Head: softmax(cls @ head_w + head_b)
    vit_gather_cls<<<(128 * 768 + 255) / 256, 256, 0, stream>>>(x, clsB);
    gemm(clsB, WhB, head_b, nullptr, lg, nullptr, Bn, 256, E, 0);
    vit_softmax256<<<Bn, 256, 0, stream>>>(lg, (float*)d_out);
}
